// Att_pooling_28913719836804
// MI455X (gfx1250) — matrix operand order repaired and checked by reading, not compiled, not run on hardware
//
#include <hip/hip_runtime.h>

#define NNODES   262144
#define DFEAT    128
#define HID      64
#define NHEADS   8
#define NGRAPHS  4096
#define DOUT1    1024
#define EPS      1e-6f
#define SLOPE    0.01f

typedef __attribute__((ext_vector_type(16))) _Float16 v16h;
typedef __attribute__((ext_vector_type(8)))  _Float16 v8h;
typedef __attribute__((ext_vector_type(8)))  float    v8f;

// A-matrix fragment (16x32 f16): lane l (g=l>>4, m=l&15) holds row m;
// halves 0-7 -> K = koff..koff+7, halves 8-15 -> K = koff+16..koff+23,
// where koff = 32*kstep + 8*g. Two contiguous 16B reads.
__device__ __forceinline__ v16h ld_afrag(const _Float16* __restrict__ row, int koff) {
  v8h lo = *(const v8h*)(row + koff);
  v8h hi = *(const v8h*)(row + koff + 16);
  v16h r;
#pragma unroll
  for (int i = 0; i < 8; ++i) { r[i] = lo[i]; r[i + 8] = hi[i]; }
  return r;
}

// B-matrix fragment (32x16 f16) from a transposed weight row (one output col):
// lane l holds col (l&15); halves 0-7 -> K = koff..koff+7, halves 8-15 -> K = koff+16..koff+23, koff = 32*kstep + 8*(l>>4).
__device__ __forceinline__ v16h ld_bfrag(const _Float16* __restrict__ row, int koff) {
  v8h lo = *(const v8h*)(row + koff);
  v8h hi = *(const v8h*)(row + koff + 16);
  v16h r;
#pragma unroll
  for (int i = 0; i < 8; ++i) { r[i] = lo[i]; r[i + 8] = hi[i]; }
  return r;
}

__device__ __forceinline__ v8f wmma16(v16h a, v16h b, v8f c) {
  return __builtin_amdgcn_wmma_f32_16x16x32_f16(false, a, false, b, (short)0, c,
                                                false, false);
}

// sum across a 16-lane half-wave (xor masks stay within the half)
__device__ __forceinline__ float lane16_sum(float v) {
  v += __shfl_xor(v, 1);
  v += __shfl_xor(v, 2);
  v += __shfl_xor(v, 4);
  v += __shfl_xor(v, 8);
  return v;
}

// ---------------- weight prep: f32 -> f16 transposed copies ----------------
__global__ __launch_bounds__(256) void prep_weights(
    const float* __restrict__ W1, const float* __restrict__ W3,
    const float* __restrict__ W4,
    _Float16* __restrict__ W1T, _Float16* __restrict__ W3T,
    _Float16* __restrict__ W4T) {
  int i = blockIdx.x * 256 + threadIdx.x;
  if (i < DFEAT * HID) {            // W1 [128][64] -> W1T [64][128]
    int k = i >> 6, n = i & 63;
    W1T[n * DFEAT + k] = (_Float16)W1[i];
  }
  if (i < DOUT1 * DFEAT) {          // W3 [1024][128] -> W3T [128][1024]
    int k = i >> 7, n = i & 127;
    W3T[n * DOUT1 + k] = (_Float16)W3[i];
  }
  if (i < DFEAT * DFEAT) {          // W4 [128][128] -> W4T [128][128]
    int k = i >> 7, n = i & 127;
    W4T[n * DFEAT + k] = (_Float16)W4[i];
  }
}

// ---------------- node phase: att = exp(LN(leaky(feat@W1+b1))@W2+b2) -------
__global__ __launch_bounds__(256) void node_kernel(
    const float* __restrict__ feat,
    const _Float16* __restrict__ W1T,
    const float* __restrict__ b1, const float* __restrict__ g1,
    const float* __restrict__ beta1,
    const float* __restrict__ W2, const float* __restrict__ b2,
    float* __restrict__ att) {
  __shared__ __align__(16) _Float16 sW1T[HID * DFEAT];   // 16 KB
  __shared__ __align__(16) _Float16 sA[8][16 * DFEAT];   // 32 KB
  __shared__ float sH[8][16 * HID];                      // 32 KB
  __shared__ float sW2[HID * NHEADS];                    //  2 KB

  const int t = threadIdx.x, wave = t >> 5, lane = t & 31;
  const int g = lane >> 4, m = lane & 15;
  const int nodeBase = blockIdx.x * 128 + wave * 16;

  for (int i = t; i < HID * DFEAT; i += 256) sW1T[i] = W1T[i];
  for (int i = t; i < HID * NHEADS; i += 256) sW2[i] = W2[i];
  for (int i = lane; i < 16 * DFEAT; i += 32)
    sA[wave][i] = (_Float16)feat[(size_t)(nodeBase + (i >> 7)) * DFEAT + (i & 127)];
  __syncthreads();

  const _Float16* arow = &sA[wave][m * DFEAT];
  float vals[4][8];  // vals[ntile][r]: H[r+8g][m+16*ntile] after bias+leaky
#pragma unroll
  for (int nt = 0; nt < 4; ++nt) {
    const _Float16* brow = &sW1T[(m + 16 * nt) * DFEAT];
    v8f c = {};
#pragma unroll
    for (int kb = 0; kb < 4; ++kb) {
      v16h a = ld_afrag(arow, 32 * kb + 8 * g);
      v16h b = ld_bfrag(brow, 32 * kb + 8 * g);
      c = wmma16(a, b, c);
    }
#pragma unroll
    for (int r = 0; r < 8; ++r) {
      float v = c[r] + b1[m + 16 * nt];
      vals[nt][r] = v > 0.f ? v : SLOPE * v;
    }
  }

  // LayerNorm per row over 64 cols (spread across 16 lanes x 4 ntiles)
#pragma unroll
  for (int r = 0; r < 8; ++r) {
    float s = vals[0][r] + vals[1][r] + vals[2][r] + vals[3][r];
    float mu = lane16_sum(s) * (1.f / 64.f);
    float q = 0.f;
#pragma unroll
    for (int nt = 0; nt < 4; ++nt) { float d = vals[nt][r] - mu; q += d * d; }
    float rstd = rsqrtf(lane16_sum(q) * (1.f / 64.f) + EPS);
    int row = r + 8 * g;
#pragma unroll
    for (int nt = 0; nt < 4; ++nt) {
      int col = m + 16 * nt;
      sH[wave][row * HID + col] = (vals[nt][r] - mu) * rstd * g1[col] + beta1[col];
    }
  }
  __syncthreads();

  // att = exp(h @ W2 + b2): 16 rows x 8 heads, 4 tasks per lane
#pragma unroll
  for (int i = 0; i < 4; ++i) {
    int id = lane + 32 * i;
    int row = id >> 3, head = id & 7;
    float s = b2[head];
    for (int c = 0; c < HID; ++c)
      s += sH[wave][row * HID + c] * sW2[c * NHEADS + head];
    att[(size_t)(nodeBase + row) * NHEADS + head] = __expf(s);
  }
}

// ---------------- pooling: out[g,h,:] = (sum att*feat) / (sum att) ---------
__device__ __forceinline__ int lower_bound_i32(const int* __restrict__ b, int n,
                                               int key) {
  int lo = 0, hi = n;
  while (lo < hi) {
    int mid = (lo + hi) >> 1;
    if (b[mid] < key) lo = mid + 1; else hi = mid;
  }
  return lo;
}

__global__ __launch_bounds__(256) void pool_kernel(
    const float* __restrict__ feat, const int* __restrict__ batch,
    const float* __restrict__ att, float* __restrict__ O1) {
  const int gid = blockIdx.x;
  const int beg = lower_bound_i32(batch, NNODES, gid);
  const int end = lower_bound_i32(batch, NNODES, gid + 1);
  const int t = threadIdx.x;
  const int d = t & 127;
  const int h0 = (t >> 7) * 4;   // threads 0-127: heads 0-3, 128-255: heads 4-7
  float acc[4] = {0.f, 0.f, 0.f, 0.f};
  float satt[4] = {0.f, 0.f, 0.f, 0.f};
  for (int i = beg; i < end; ++i) {
    float f = feat[(size_t)i * DFEAT + d];
#pragma unroll
    for (int j = 0; j < 4; ++j) {
      float a = att[(size_t)i * NHEADS + h0 + j];
      acc[j] += a * f;
      satt[j] += a;
    }
  }
#pragma unroll
  for (int j = 0; j < 4; ++j) {
    float o = (end > beg) ? acc[j] / satt[j] : 0.f;
    O1[(size_t)gid * DOUT1 + (h0 + j) * DFEAT + d] = o;
  }
}

// ---------------- output block: two GEMM + leaky + LN stages ---------------
__global__ __launch_bounds__(256) void out_kernel(
    const float* __restrict__ O1, const _Float16* __restrict__ W3T,
    const float* __restrict__ b3, const float* __restrict__ g3,
    const float* __restrict__ beta3,
    const _Float16* __restrict__ W4T,
    const float* __restrict__ b4, const float* __restrict__ g4,
    const float* __restrict__ beta4,
    float* __restrict__ out) {
  __shared__ __align__(16) _Float16 sA[16 * DOUT1];      // 32 KB
  __shared__ float sX[16 * DFEAT];                       //  8 KB
  __shared__ __align__(16) _Float16 sA2[16 * DFEAT];     //  4 KB
  __shared__ __align__(16) _Float16 sW4[DFEAT * DFEAT];  // 32 KB

  const int t = threadIdx.x, wave = t >> 5, lane = t & 31;
  const int g = lane >> 4, m = lane & 15;
  const int row0 = blockIdx.x * 16;

  for (int i = t; i < 16 * DOUT1; i += 256)
    sA[i] = (_Float16)O1[(size_t)(row0 + (i >> 10)) * DOUT1 + (i & 1023)];
  for (int i = t; i < DFEAT * DFEAT; i += 256) sW4[i] = W4T[i];
  __syncthreads();

  // GEMM1: [16,1024]@[1024,128]; wave w owns output cols 16w..16w+15
  {
    const _Float16* arow = &sA[m * DOUT1];
    const _Float16* brow = &W3T[(size_t)(m + 16 * wave) * DOUT1];
    v8f c = {};
    for (int kb = 0; kb < 32; ++kb) {
      v16h a = ld_afrag(arow, 32 * kb + 8 * g);
      v16h b = ld_bfrag(brow, 32 * kb + 8 * g);
      c = wmma16(a, b, c);
    }
#pragma unroll
    for (int r = 0; r < 8; ++r) {
      int row = r + 8 * g, col = m + 16 * wave;
      float v = c[r] + b3[col];
      sX[row * DFEAT + col] = v > 0.f ? v : SLOPE * v;
    }
  }
  __syncthreads();

  // LN over 128 cols; wave handles rows 2*wave + g (16 lanes per row)
  {
    int row = 2 * wave + g;
    float vals[8];
#pragma unroll
    for (int j = 0; j < 8; ++j) vals[j] = sX[row * DFEAT + m + 16 * j];
    float s = 0.f;
#pragma unroll
    for (int j = 0; j < 8; ++j) s += vals[j];
    float mu = lane16_sum(s) * (1.f / 128.f);
    float q = 0.f;
#pragma unroll
    for (int j = 0; j < 8; ++j) { float d = vals[j] - mu; q += d * d; }
    float rstd = rsqrtf(lane16_sum(q) * (1.f / 128.f) + EPS);
#pragma unroll
    for (int j = 0; j < 8; ++j) {
      int col = m + 16 * j;
      sA2[row * DFEAT + col] =
          (_Float16)((vals[j] - mu) * rstd * g3[col] + beta3[col]);
    }
  }
  __syncthreads();

  // GEMM2: [16,128]@[128,128]
  {
    const _Float16* arow = &sA2[m * DFEAT];
    const _Float16* brow = &sW4[(m + 16 * wave) * DFEAT];
    v8f c = {};
#pragma unroll
    for (int kb = 0; kb < 4; ++kb) {
      v16h a = ld_afrag(arow, 32 * kb + 8 * g);
      v16h b = ld_bfrag(brow, 32 * kb + 8 * g);
      c = wmma16(a, b, c);
    }
#pragma unroll
    for (int r = 0; r < 8; ++r) {
      int row = r + 8 * g, col = m + 16 * wave;
      float v = c[r] + b4[col];
      sX[row * DFEAT + col] = v > 0.f ? v : SLOPE * v;
    }
  }
  __syncthreads();

  // final LN + store
  {
    int row = 2 * wave + g;
    float vals[8];
#pragma unroll
    for (int j = 0; j < 8; ++j) vals[j] = sX[row * DFEAT + m + 16 * j];
    float s = 0.f;
#pragma unroll
    for (int j = 0; j < 8; ++j) s += vals[j];
    float mu = lane16_sum(s) * (1.f / 128.f);
    float q = 0.f;
#pragma unroll
    for (int j = 0; j < 8; ++j) { float d = vals[j] - mu; q += d * d; }
    float rstd = rsqrtf(lane16_sum(q) * (1.f / 128.f) + EPS);
#pragma unroll
    for (int j = 0; j < 8; ++j) {
      int col = m + 16 * j;
      out[(size_t)(row0 + row) * DFEAT + col] =
          (vals[j] - mu) * rstd * g4[col] + beta4[col];
    }
  }
}

extern "C" void kernel_launch(void* const* d_in, const int* in_sizes, int n_in,
                              void* d_out, int out_size, void* d_ws,
                              size_t ws_size, hipStream_t stream) {
  (void)in_sizes; (void)n_in; (void)out_size; (void)ws_size;
  const float* feat = (const float*)d_in[0];
  const int* batch  = (const int*)d_in[1];
  const float* W1 = (const float*)d_in[2];
  const float* b1 = (const float*)d_in[3];
  const float* g1 = (const float*)d_in[4];
  const float* be1 = (const float*)d_in[5];
  const float* W2 = (const float*)d_in[6];
  const float* b2 = (const float*)d_in[7];
  const float* W3 = (const float*)d_in[8];
  const float* b3 = (const float*)d_in[9];
  const float* g3 = (const float*)d_in[10];
  const float* be3 = (const float*)d_in[11];
  const float* W4 = (const float*)d_in[12];
  const float* b4 = (const float*)d_in[13];
  const float* g4 = (const float*)d_in[14];
  const float* be4 = (const float*)d_in[15];

  char* ws = (char*)d_ws;
  float* att = (float*)ws;                               //  8 MB  [N][8]
  float* O1 = (float*)(ws + (size_t)8 * 1024 * 1024);    // 16 MB  [B][1024]
  _Float16* W1T = (_Float16*)(ws + 25165824);            // 16 KB
  _Float16* W3T = (_Float16*)(ws + 25165824 + 16384);    // 256 KB
  _Float16* W4T = (_Float16*)(ws + 25165824 + 16384 + 262144);  // 32 KB

  prep_weights<<<512, 256, 0, stream>>>(W1, W3, W4, W1T, W3T, W4T);
  node_kernel<<<NNODES / 128, 256, 0, stream>>>(feat, W1T, b1, g1, be1, W2, b2,
                                                att);
  pool_kernel<<<NGRAPHS, 256, 0, stream>>>(feat, batch, att, O1);
  out_kernel<<<NGRAPHS / 16, 256, 0, stream>>>(O1, W3T, b3, g3, be3, W4T, b4,
                                               g4, be4, (float*)d_out);
}
